// SpectralGraphConv_41927470743646
// MI455X (gfx1250) — matrix operand order repaired and checked by reading, not compiled, not run on hardware
//
#include <hip/hip_runtime.h>

typedef __bf16 bf16_t;
typedef bf16_t v16bf __attribute__((ext_vector_type(16)));
typedef bf16_t v8bf  __attribute__((ext_vector_type(8)));
typedef bf16_t v4bf  __attribute__((ext_vector_type(4)));
typedef float  v8f   __attribute__((ext_vector_type(8)));
typedef float  v4f   __attribute__((ext_vector_type(4)));

#define NN 8192      // nodes
#define DD 256       // input dim (= per-T feature count)
#define KORD 5       // Chebyshev order
#define FF 256       // output features
#define LDST 40      // LDS row stride in bf16 (32 data + 8 pad -> 80B)

// ---- CDNA5 async global->LDS helpers (ASYNCcnt-tracked) --------------------
__device__ __forceinline__ unsigned lds_off32(const void* p) {
  // flat aperture: LDS offset is the pointer's low 32 bits
  return (unsigned)(unsigned long long)p;
}
__device__ __forceinline__ void wait_async0() {
  asm volatile("s_wait_asynccnt 0" ::: "memory");
}

// ---------------------------------------------------------------------------
// elementwise fp32 -> bf16 (for L)
// ---------------------------------------------------------------------------
__global__ void cvt_f32_to_bf16(const float* __restrict__ src,
                                bf16_t* __restrict__ dst, long long n) {
  long long i = ((long long)blockIdx.x * blockDim.x + threadIdx.x) * 4;
  long long stride = (long long)gridDim.x * blockDim.x * 4;
  for (; i < n; i += stride) {
    v4f v = *(const v4f*)(src + i);
    v4bf o;
    o[0] = (bf16_t)v[0]; o[1] = (bf16_t)v[1];
    o[2] = (bf16_t)v[2]; o[3] = (bf16_t)v[3];
    *(v4bf*)(dst + i) = o;
  }
}

// x -> feats_f32 slot0 (copy) + transposed bf16 slot0 [DD][NN]
__global__ void cvt_x(const float* __restrict__ x,
                      float* __restrict__ f32_0,
                      bf16_t* __restrict__ tbt0) {
  int tid = blockIdx.x * blockDim.x + threadIdx.x;   // 0 .. NN*DD-1
  int n = tid >> 8;
  int d = tid & 255;
  float v = x[tid];
  f32_0[tid] = v;
  tbt0[(size_t)d * NN + n] = (bf16_t)v;
}

// W [KORD*DD][FF] row-major -> Wt bf16 [FF][KORD*DD]
__global__ void cvt_w(const float* __restrict__ W, bf16_t* __restrict__ Wt) {
  int tid = blockIdx.x * blockDim.x + threadIdx.x;   // 0 .. KORD*DD*FF-1
  int kd = tid >> 8;
  int f  = tid & 255;
  Wt[(size_t)f * (KORD * DD) + kd] = (bf16_t)W[tid];
}

// ---------------------------------------------------------------------------
// Chebyshev GEMM:  out = alpha * (A @ T) + beta * Cprev
// Double-buffered async global->LDS staging (GLOBAL_LOAD_ASYNC_TO_LDS_B128),
// one barrier per K-step, 4x v_wmma_f32_16x16x32_bf16 per step per wave.
//   A      : bf16 [NN][NN] row-major
//   BT     : bf16 [DD][NN]  (T_{k-1} transposed)
//   Cprev  : f32  [NN][DD]  (T_{k-2}, only if HAS_PREV)
//   Cf32   : f32  [NN][DD]  (T_k out)
//   CTbf   : bf16 [DD][NN]  (T_k transposed, next pass's B)
// ---------------------------------------------------------------------------
template <bool HAS_PREV>
__global__ void __launch_bounds__(128)
cheb_gemm(const bf16_t* __restrict__ A,
          const bf16_t* __restrict__ BT,
          const float*  __restrict__ Cprev,
          float*        __restrict__ Cf32,
          bf16_t*       __restrict__ CTbf,
          float alpha, float beta) {
  __shared__ bf16_t ldsA[2][64 * LDST];
  __shared__ bf16_t ldsB[2][64 * LDST];

  const int t     = threadIdx.x;
  const int lane  = t & 31;
  const int wave  = t >> 5;        // 0..3
  const int waveM = wave & 1;
  const int waveN = wave >> 1;
  const int l15   = lane & 15;
  const int lhalf = lane >> 4;     // 0/1
  const int kbA   = lhalf * 8;     // A frag: K base 0/8 (hi half at +16)
  const int kbB   = lhalf * 8;     // B frag: K base 0/8 (hi half at +16)

  const int m0 = blockIdx.x * 64;
  const int n0 = blockIdx.y * 64;

  // per-thread staging chunks: tile = 256 x 16B chunks, 2 per thread per tile
  const int c0 = t, c1 = t + 128;
  const int r0 = c0 >> 2, x0 = (c0 & 3) * 8;
  const int r1 = c1 >> 2, x1 = (c1 & 3) * 8;
  const bf16_t* pA0 = A  + (size_t)(m0 + r0) * NN + x0;
  const bf16_t* pA1 = A  + (size_t)(m0 + r1) * NN + x1;
  const bf16_t* pB0 = BT + (size_t)(n0 + r0) * NN + x0;
  const bf16_t* pB1 = BT + (size_t)(n0 + r1) * NN + x1;
  unsigned aoff0[2], aoff1[2], boff0[2], boff1[2];
#pragma unroll
  for (int b = 0; b < 2; ++b) {
    aoff0[b] = lds_off32(&ldsA[b][r0 * LDST + x0]);
    aoff1[b] = lds_off32(&ldsA[b][r1 * LDST + x1]);
    boff0[b] = lds_off32(&ldsB[b][r0 * LDST + x0]);
    boff1[b] = lds_off32(&ldsB[b][r1 * LDST + x1]);
  }

  auto stage = [&](int buf, int kk) {
    asm volatile(
        "global_load_async_to_lds_b128 %0, %4, off\n\t"
        "global_load_async_to_lds_b128 %1, %5, off\n\t"
        "global_load_async_to_lds_b128 %2, %6, off\n\t"
        "global_load_async_to_lds_b128 %3, %7, off"
        :
        : "v"(aoff0[buf]), "v"(aoff1[buf]), "v"(boff0[buf]), "v"(boff1[buf]),
          "v"(pA0 + kk), "v"(pA1 + kk), "v"(pB0 + kk), "v"(pB1 + kk)
        : "memory");
  };

  const v8f zero = {0.f, 0.f, 0.f, 0.f, 0.f, 0.f, 0.f, 0.f};
  v8f acc[2][2];
  acc[0][0] = zero; acc[0][1] = zero; acc[1][0] = zero; acc[1][1] = zero;

  stage(0, 0);
  for (int kk = 0; kk < NN; kk += 32) {
    const int cur = (kk >> 5) & 1;
    wait_async0();            // my async writes into buf[cur] have landed
    __syncthreads();          // everyone's writes visible; prior reads done
    if (kk + 32 < NN) stage(cur ^ 1, kk + 32);  // overlap with compute

    const bf16_t* LA = ldsA[cur];
    const bf16_t* LB = ldsB[cur];
    v16bf aF[2], bF[2];
#pragma unroll
    for (int mt = 0; mt < 2; ++mt) {
      int row = waveM * 32 + mt * 16 + l15;
      v8bf lo = *(const v8bf*)&LA[row * LDST + kbA];
      v8bf hi = *(const v8bf*)&LA[row * LDST + kbA + 16];
#pragma unroll
      for (int i = 0; i < 8; ++i) { aF[mt][i] = lo[i]; aF[mt][8 + i] = hi[i]; }
    }
#pragma unroll
    for (int nt = 0; nt < 2; ++nt) {
      int col = waveN * 32 + nt * 16 + l15;
      v8bf lo = *(const v8bf*)&LB[col * LDST + kbB];
      v8bf hi = *(const v8bf*)&LB[col * LDST + kbB + 16];
#pragma unroll
      for (int i = 0; i < 8; ++i) { bF[nt][i] = lo[i]; bF[nt][8 + i] = hi[i]; }
    }
#pragma unroll
    for (int mt = 0; mt < 2; ++mt)
#pragma unroll
      for (int nt = 0; nt < 2; ++nt)
        acc[mt][nt] = __builtin_amdgcn_wmma_f32_16x16x32_bf16(
            false, aF[mt], false, bF[nt], (short)0, acc[mt][nt], false, false);
  }

  // epilogue: V = alpha*acc (+ beta*Cprev); f32 row-major + bf16 transposed
#pragma unroll
  for (int mt = 0; mt < 2; ++mt) {
#pragma unroll
    for (int nt = 0; nt < 2; ++nt) {
      int mBase = m0 + waveM * 32 + mt * 16 + lhalf * 8;
      int nCol  = n0 + waveN * 32 + nt * 16 + l15;
      float prev[8];
      if (HAS_PREV) {
#pragma unroll
        for (int j = 0; j < 8; ++j)
          prev[j] = Cprev[(size_t)(mBase + j) * DD + nCol];
      }
      v8bf packed;
#pragma unroll
      for (int j = 0; j < 8; ++j) {
        float v = alpha * acc[mt][nt][j];
        if (HAS_PREV) v += beta * prev[j];
        Cf32[(size_t)(mBase + j) * DD + nCol] = v;
        packed[j] = (bf16_t)v;
      }
      *(v8bf*)(CTbf + (size_t)nCol * NN + mBase) = packed;  // 16B contiguous
    }
  }
}

// ---------------------------------------------------------------------------
// Final GEMM: out = relu(concat(feats) @ W + b)
//   feats : f32 [KORD][NN][DD]; K dim = KORD*DD = 1280
//   WT    : bf16 [FF][KORD*DD]
// ---------------------------------------------------------------------------
__global__ void __launch_bounds__(128)
final_gemm(const float* __restrict__ feats,
           const bf16_t* __restrict__ WT,
           const float*  __restrict__ bias,
           float*        __restrict__ out) {
  __shared__ bf16_t ldsA[64 * LDST];
  __shared__ bf16_t ldsB[64 * LDST];

  const int t     = threadIdx.x;
  const int lane  = t & 31;
  const int wave  = t >> 5;
  const int waveM = wave & 1;
  const int waveN = wave >> 1;
  const int l15   = lane & 15;
  const int lhalf = lane >> 4;
  const int kbA   = lhalf * 8;
  const int kbB   = lhalf * 8;

  const int m0 = blockIdx.x * 64;
  const int n0 = blockIdx.y * 64;

  const v8f zero = {0.f, 0.f, 0.f, 0.f, 0.f, 0.f, 0.f, 0.f};
  v8f acc[2][2];
  acc[0][0] = zero; acc[0][1] = zero; acc[1][0] = zero; acc[1][1] = zero;

  for (int kk = 0; kk < KORD * DD; kk += 32) {
    int ck = kk >> 8;        // which T_k chunk
    int d0 = kk & 255;       // column offset inside chunk (multiple of 32)
    const float* Asrc = feats + (size_t)ck * NN * DD + d0;

    __syncthreads();
    {  // stage A: 64 rows x 32 f32 -> bf16; thread: row t>>1, half t&1
      int r = t >> 1;
      int h = t & 1;
      const float* p = Asrc + (size_t)(m0 + r) * DD + h * 16;
      v4f f0 = *(const v4f*)(p + 0);
      v4f f1 = *(const v4f*)(p + 4);
      v4f f2 = *(const v4f*)(p + 8);
      v4f f3 = *(const v4f*)(p + 12);
      v8bf o0, o1;
#pragma unroll
      for (int i = 0; i < 4; ++i) {
        o0[i]     = (bf16_t)f0[i];
        o0[4 + i] = (bf16_t)f1[i];
        o1[i]     = (bf16_t)f2[i];
        o1[4 + i] = (bf16_t)f3[i];
      }
      *(v8bf*)&ldsA[r * LDST + h * 16]     = o0;
      *(v8bf*)&ldsA[r * LDST + h * 16 + 8] = o1;
    }
#pragma unroll
    for (int j = 0; j < 2; ++j) {  // stage B from WT, contiguous 16B chunks
      int c   = t + j * 128;
      int r   = c >> 2;
      int c16 = c & 3;
      *(v8bf*)&ldsB[r * LDST + c16 * 8] =
          *(const v8bf*)(WT + (size_t)(n0 + r) * (KORD * DD) + kk + c16 * 8);
    }
    __syncthreads();

    v16bf aF[2], bF[2];
#pragma unroll
    for (int mt = 0; mt < 2; ++mt) {
      int row = waveM * 32 + mt * 16 + l15;
      v8bf lo = *(const v8bf*)&ldsA[row * LDST + kbA];
      v8bf hi = *(const v8bf*)&ldsA[row * LDST + kbA + 16];
#pragma unroll
      for (int i = 0; i < 8; ++i) { aF[mt][i] = lo[i]; aF[mt][8 + i] = hi[i]; }
    }
#pragma unroll
    for (int nt = 0; nt < 2; ++nt) {
      int col = waveN * 32 + nt * 16 + l15;
      v8bf lo = *(const v8bf*)&ldsB[col * LDST + kbB];
      v8bf hi = *(const v8bf*)&ldsB[col * LDST + kbB + 16];
#pragma unroll
      for (int i = 0; i < 8; ++i) { bF[nt][i] = lo[i]; bF[nt][8 + i] = hi[i]; }
    }
#pragma unroll
    for (int mt = 0; mt < 2; ++mt)
#pragma unroll
      for (int nt = 0; nt < 2; ++nt)
        acc[mt][nt] = __builtin_amdgcn_wmma_f32_16x16x32_bf16(
            false, aF[mt], false, bF[nt], (short)0, acc[mt][nt], false, false);
  }

  // epilogue: bias + relu, f32 row-major
#pragma unroll
  for (int mt = 0; mt < 2; ++mt) {
#pragma unroll
    for (int nt = 0; nt < 2; ++nt) {
      int mBase = m0 + waveM * 32 + mt * 16 + lhalf * 8;
      int nCol  = n0 + waveN * 32 + nt * 16 + l15;
      float bv = bias[nCol];
#pragma unroll
      for (int j = 0; j < 8; ++j) {
        float v = acc[mt][nt][j] + bv;
        out[(size_t)(mBase + j) * FF + nCol] = v > 0.f ? v : 0.f;
      }
    }
  }
}

// ---------------------------------------------------------------------------
// launch
// ---------------------------------------------------------------------------
extern "C" void kernel_launch(void* const* d_in, const int* in_sizes, int n_in,
                              void* d_out, int out_size, void* d_ws, size_t ws_size,
                              hipStream_t stream) {
  const float* L = (const float*)d_in[0];   // [NN*NN]
  const float* x = (const float*)d_in[1];   // [NN*DD]
  const float* W = (const float*)d_in[2];   // [KORD*DD*FF]
  const float* b = (const float*)d_in[3];   // [FF]
  float* out = (float*)d_out;

  // workspace layout (bytes):
  //   Lh    bf16 [NN][NN]            134,217,728  (fits in 192MB L2 after pass 1)
  //   feats f32  [KORD][NN][DD]       41,943,040
  //   TbT   bf16 [KORD][DD][NN]       20,971,520
  //   Wt    bf16 [FF][KORD*DD]           655,360
  constexpr size_t LH_BYTES    = (size_t)NN * NN * 2;
  constexpr size_t FEAT_ELEMS  = (size_t)NN * DD;           // per slot
  constexpr size_t FEATS_BYTES = (size_t)KORD * FEAT_ELEMS * 4;
  constexpr size_t TBT_BYTES   = (size_t)KORD * FEAT_ELEMS * 2;

  char* ws = (char*)d_ws;
  bf16_t* Lh    = (bf16_t*)ws;
  float*  feats = (float*)(ws + LH_BYTES);
  bf16_t* TbT   = (bf16_t*)(ws + LH_BYTES + FEATS_BYTES);
  bf16_t* Wt    = (bf16_t*)(ws + LH_BYTES + FEATS_BYTES + TBT_BYTES);

  cvt_f32_to_bf16<<<4096, 256, 0, stream>>>(L, Lh, (long long)NN * NN);
  cvt_x<<<(NN * DD) / 256, 256, 0, stream>>>(x, feats, TbT);
  cvt_w<<<(KORD * DD * FF) / 256, 256, 0, stream>>>(W, Wt);

  dim3 grid(NN / 64, DD / 64);  // 128 x 4

  // k = 1: T1 = L @ x
  cheb_gemm<false><<<grid, 128, 0, stream>>>(Lh, TbT, nullptr,
                                             feats + FEAT_ELEMS,
                                             TbT + FEAT_ELEMS, 1.0f, 0.0f);
  // k = 2..4: T_k = 2 L T_{k-1} - T_{k-2}
  for (int k = 2; k < KORD; ++k) {
    cheb_gemm<true><<<grid, 128, 0, stream>>>(Lh,
                                              TbT + (size_t)(k - 1) * FEAT_ELEMS,
                                              feats + (size_t)(k - 2) * FEAT_ELEMS,
                                              feats + (size_t)k * FEAT_ELEMS,
                                              TbT + (size_t)k * FEAT_ELEMS,
                                              2.0f, -1.0f);
  }

  final_gemm<<<grid, 128, 0, stream>>>(feats, Wt, b, out);
}
